// FacialGNN_88768384073894
// MI455X (gfx1250) — matrix operand order repaired and checked by reading, not compiled, not run on hardware
//
#include <hip/hip_runtime.h>

// ---------------- problem constants ----------------
constexpr int NN  = 69632;   // nodes
constexpr int EE  = 557056;  // edges
constexpr int CC  = 3;       // input channels
constexpr int PP  = 16;      // patch
constexpr int KK  = 8;       // conv kernels
constexpr int FD  = 64;      // feature dim
constexpr int GH  = 128;     // gcn hidden
constexpr int GG  = 1024;    // graphs
constexpr int FLAT = 512;    // K * 8 * 8

typedef __attribute__((ext_vector_type(16))) __bf16 v16bf;
typedef __attribute__((ext_vector_type(8)))  float  v8f;

union ABFrag {
    unsigned int   u[8];
    unsigned short h[16];
    v16bf          v;
};

__device__ __forceinline__ unsigned short f32_to_bf16(float f) {
    union { float f; unsigned int u; } x; x.f = f;
    unsigned int r = x.u + 0x7FFFu + ((x.u >> 16) & 1u);  // RNE
    return (unsigned short)(r >> 16);
}

// ============================================================
// Kernel 1: conv3x3(SAME) + bias + ReLU + maxpool2x2 as implicit
// GEMM on the matrix pipe. One block (512 thr = 16 waves) per node.
// im2col: A[pixel(256) x K(27->32)] @ B[K x 8(->16 kernels)]
// Wave w computes output row py = w. All im2col LDS reads use
// compile-time offsets (branch on lane-half, reconverges pre-WMMA).
// ============================================================
__global__ __launch_bounds__(512) void conv_pool_kernel(
    const float* __restrict__ x, const float* __restrict__ conv_w,
    const float* __restrict__ conv_b, unsigned short* __restrict__ pooled)
{
    __shared__ unsigned short Xs[CC][PP + 2][PP + 2]; // zero-padded patch, bf16
    __shared__ unsigned short Ws[16][36];             // Ws[n][kk], padded pitch
    __shared__ float          Hs[16][KK][8];          // horiz max [py][k][px/2]

    const int node = blockIdx.x;
    const int tid  = threadIdx.x;

    for (int i = tid; i < CC * 18 * 18; i += 512)
        ((unsigned short*)Xs)[i] = 0;
    for (int i = tid; i < 16 * 32; i += 512) {
        int n = i >> 5, kk = i & 31;
        float w = (kk < 27 && n < KK) ? conv_w[n * 27 + kk] : 0.f;
        Ws[n][kk] = f32_to_bf16(w);
    }
    __syncthreads();
    for (int i = tid; i < CC * 256; i += 512) {
        int c = i >> 8, p = i & 255;
        Xs[c][(p >> 4) + 1][(p & 15) + 1] =
            f32_to_bf16(x[((size_t)node * CC + c) * 256 + p]);
    }
    __syncthreads();

    const int lane = tid & 31;
    const int wave = tid >> 5;      // = output row py (0..15)
    const int half = lane >> 4;
    const int nidx = lane & 15;     // B column (kernel id; 8..15 padding)
    const int px   = lane & 15;     // A row M = pixel column
    const int py   = wave;

    // B fragment: table order (e<8: K = 8*half + e; e>=8: K = 16 + 8*half + e-8) -> u32 loads
    ABFrag bf;
    #pragma unroll
    for (int q = 0; q < 8; ++q)
        bf.u[q] = *(const unsigned int*)&Ws[nidx][((q & 4) << 2) + half * 8 + ((q & 3) << 1)];

    // A fragment: all (c,dy,dx) compile-time; only (py,px) base is runtime.
    ABFrag af;
    #define LDX(c, dy, dx) Xs[c][py + (dy)][px + (dx)]
    if (half == 0) {
        // h[0..7] = kk 0..7 ; h[8..15] = kk 16..23
        af.h[0]  = LDX(0,0,0); af.h[1]  = LDX(0,0,1);
        af.h[2]  = LDX(0,0,2); af.h[3]  = LDX(0,1,0);
        af.h[4]  = LDX(0,1,1); af.h[5]  = LDX(0,1,2);
        af.h[6]  = LDX(0,2,0); af.h[7]  = LDX(0,2,1);
        af.h[8]  = LDX(1,2,1); af.h[9]  = LDX(1,2,2);
        af.h[10] = LDX(2,0,0); af.h[11] = LDX(2,0,1);
        af.h[12] = LDX(2,0,2); af.h[13] = LDX(2,1,0);
        af.h[14] = LDX(2,1,1); af.h[15] = LDX(2,1,2);
    } else {
        // h[0..7] = kk 8..15 ; h[8..10] = kk 24..26 ; h[11..15] = 0 pad
        af.h[0]  = LDX(0,2,2); af.h[1]  = LDX(1,0,0);
        af.h[2]  = LDX(1,0,1); af.h[3]  = LDX(1,0,2);
        af.h[4]  = LDX(1,1,0); af.h[5]  = LDX(1,1,1);
        af.h[6]  = LDX(1,1,2); af.h[7]  = LDX(1,2,0);
        af.h[8]  = LDX(2,2,0); af.h[9]  = LDX(2,2,1);
        af.h[10] = LDX(2,2,2); af.h[11] = 0;
        af.h[12] = 0; af.h[13] = 0; af.h[14] = 0; af.h[15] = 0;
    }
    #undef LDX

    v8f acc = {};
    acc = __builtin_amdgcn_wmma_f32_16x16x32_bf16(
        false, af.v, false, bf.v, (short)0, acc, false, false);

    // d[j] = conv(py, px = j + 8*half, kernel = nidx); bias+relu+hmax
    if (nidx < KK) {
        float b = conv_b[nidx];
        #pragma unroll
        for (int a = 0; a < 4; ++a) {
            float v0 = fmaxf(acc[2 * a]     + b, 0.f);
            float v1 = fmaxf(acc[2 * a + 1] + b, 0.f);
            Hs[py][nidx][half * 4 + a] = fmaxf(v0, v1);
        }
    }
    __syncthreads();

    // vertical max + store: 512 pooled outputs, flatten order k*64 + pr*8 + pc
    int o  = tid;
    int k  = o >> 6, pr = (o >> 3) & 7, pc = o & 7;
    float m = fmaxf(Hs[2 * pr][k][pc], Hs[2 * pr + 1][k][pc]);
    pooled[(size_t)node * FLAT + o] = f32_to_bf16(m);
}

// ============================================================
// Kernel 2: feats = relu(pooled[N,512] @ lin1_w[512,64] + b)
// Block: 256 thr = 8 waves -> 32 rows x 64 cols (2x4 wave tiles).
// K staged in chunks of 128 (4 WMMAs per stage, 4 outer iters).
// LDS pitch 136 halfs (272 B) -> conflict-free lane-indexed rows.
// ============================================================
__global__ __launch_bounds__(256) void lin1_kernel(
    const unsigned short* __restrict__ pooled, const float* __restrict__ w,
    const float* __restrict__ b, unsigned short* __restrict__ feats)
{
    __shared__ unsigned short As[32][136];  // [row][kk]
    __shared__ unsigned short Bs[64][136];  // [col][kk] (transposed)

    const int row0 = blockIdx.x * 32;
    const int tid  = threadIdx.x;
    const int lane = tid & 31, wave = tid >> 5;
    const int rt = wave >> 2, ct = wave & 3;
    const int half = lane >> 4, nl = lane & 15;
    const int m = rt * 16 + nl, coln = ct * 16 + nl;

    v8f acc = {};
    for (int k0 = 0; k0 < FLAT; k0 += 128) {
        __syncthreads();
        {   // stage A: 32x128 bf16; 16 contiguous elems per thread
            int r  = tid >> 3;
            int cb = (tid & 7) << 4;
            const ushort4* src =
                (const ushort4*)(pooled + (size_t)(row0 + r) * FLAT + k0 + cb);
            ushort4* dst = (ushort4*)&As[r][cb];
            #pragma unroll
            for (int q = 0; q < 4; ++q) dst[q] = src[q];
        }
        {   // stage B transposed: float4 rows -> bf16 cols
            int kb   = tid >> 4;          // 0..15
            int col4 = (tid & 15) << 2;   // 0..60
            #pragma unroll
            for (int kk = 0; kk < 128; kk += 16) {
                float4 wv = *(const float4*)&w[(size_t)(k0 + kb + kk) * FD + col4];
                Bs[col4 + 0][kb + kk] = f32_to_bf16(wv.x);
                Bs[col4 + 1][kb + kk] = f32_to_bf16(wv.y);
                Bs[col4 + 2][kb + kk] = f32_to_bf16(wv.z);
                Bs[col4 + 3][kb + kk] = f32_to_bf16(wv.w);
            }
        }
        __syncthreads();

        #pragma unroll
        for (int ks = 0; ks < 128; ks += 32) {
            ABFrag af, bfr;
            #pragma unroll
            for (int j = 0; j < 8; ++j) {
                int kb = ks + ((j & 4) << 2) + half * 8 + ((j & 3) << 1);
                af.u[j] = *(const unsigned int*)&As[m][kb];
            }
            #pragma unroll
            for (int q = 0; q < 8; ++q)
                bfr.u[q] = *(const unsigned int*)&Bs[coln][ks + ((q & 4) << 2) + half * 8 + ((q & 3) << 1)];
            acc = __builtin_amdgcn_wmma_f32_16x16x32_bf16(
                false, af.v, false, bfr.v, (short)0, acc, false, false);
        }
    }

    float bias = b[coln];
    #pragma unroll
    for (int j = 0; j < 8; ++j) {
        int r = row0 + rt * 16 + j + 8 * half;
        feats[(size_t)r * FD + coln] = f32_to_bf16(fmaxf(acc[j] + bias, 0.f));
    }
}

// ============================================================
// Kernel 3: xw = feats[N,64] @ gcn_w[64,128]   (fp32 out, no bias)
// Block: 512 thr = 16 waves -> 32 rows x 128 cols, K = 2 wmma steps.
// LDS pitch 72 halfs (144 B) -> conflict-free lane-indexed rows.
// ============================================================
__global__ __launch_bounds__(512) void gcnxw_kernel(
    const unsigned short* __restrict__ feats, const float* __restrict__ gw,
    float* __restrict__ xw)
{
    __shared__ unsigned short As[32][72];
    __shared__ unsigned short Bs[GH][72];  // [col][kk]

    const int row0 = blockIdx.x * 32;
    const int tid  = threadIdx.x;
    const int lane = tid & 31, wave = tid >> 5;
    const int rt = wave >> 3, ct = wave & 7;
    const int half = lane >> 4, nl = lane & 15;
    const int m = rt * 16 + nl, coln = ct * 16 + nl;

    {   // stage A whole (32x64), 4 contiguous elems per thread
        int r  = tid >> 4;
        int c4 = (tid & 15) << 2;
        *(ushort4*)&As[r][c4] =
            *(const ushort4*)(feats + (size_t)(row0 + r) * FD + c4);
    }
    {   // stage B transposed: float4 rows -> bf16 cols
        int kb   = tid >> 5;          // 0..15
        int col4 = (tid & 31) << 2;   // 0..124
        #pragma unroll
        for (int kk = 0; kk < 64; kk += 16) {
            float4 wv = *(const float4*)&gw[(size_t)(kb + kk) * GH + col4];
            Bs[col4 + 0][kb + kk] = f32_to_bf16(wv.x);
            Bs[col4 + 1][kb + kk] = f32_to_bf16(wv.y);
            Bs[col4 + 2][kb + kk] = f32_to_bf16(wv.z);
            Bs[col4 + 3][kb + kk] = f32_to_bf16(wv.w);
        }
    }
    __syncthreads();

    v8f acc = {};
    #pragma unroll
    for (int k0 = 0; k0 < FD; k0 += 32) {
        ABFrag af, bfr;
        #pragma unroll
        for (int j = 0; j < 8; ++j) {
            int kb = k0 + ((j & 4) << 2) + half * 8 + ((j & 3) << 1);
            af.u[j] = *(const unsigned int*)&As[m][kb];
        }
        #pragma unroll
        for (int q = 0; q < 8; ++q)
            bfr.u[q] = *(const unsigned int*)&Bs[coln][k0 + ((q & 4) << 2) + half * 8 + ((q & 3) << 1)];
        acc = __builtin_amdgcn_wmma_f32_16x16x32_bf16(
            false, af.v, false, bfr.v, (short)0, acc, false, false);
    }

    #pragma unroll
    for (int j = 0; j < 8; ++j) {
        int r = row0 + rt * 16 + j + 8 * half;
        xw[(size_t)r * GH + coln] = acc[j];
    }
}

// ============================================================
// Graph stages
// ============================================================
__global__ void init_kernel(float* deg, float* sums, float* cnt) {
    int i = blockIdx.x * blockDim.x + threadIdx.x;
    if (i < NN) deg[i] = 1.0f;           // self-loop
    if (i < GG * GH) sums[i] = 0.f;
    if (i < GG) cnt[i] = 0.f;
}

__global__ void deg_edges_kernel(const int* __restrict__ ei, float* deg) {
    int e = blockIdx.x * blockDim.x + threadIdx.x;
    if (e < EE) atomicAdd(&deg[ei[EE + e]], 1.0f);   // col = target
}

__global__ void dis_kernel(const float* __restrict__ deg, float* dis) {
    int i = blockIdx.x * blockDim.x + threadIdx.x;
    if (i < NN) { float d = deg[i]; dis[i] = d > 0.f ? rsqrtf(d) : 0.f; }
}

// agg init = self-loop term: xw[i] * dis[i]^2
__global__ void selfloop_kernel(const float* __restrict__ xw,
                                const float* __restrict__ dis,
                                float* __restrict__ agg) {
    int i = blockIdx.x * blockDim.x + threadIdx.x;  // N*GH threads
    int n = i >> 7;
    float s = dis[n];
    agg[i] = xw[i] * s * s;
}

// one wave per edge; lane gathers float4 of xw[row], atomically adds to agg[col]
__global__ __launch_bounds__(256) void scatter_kernel(
    const int* __restrict__ ei, const float* __restrict__ xw,
    const float* __restrict__ dis, float* __restrict__ agg)
{
    int gid  = blockIdx.x * blockDim.x + threadIdx.x;
    int e    = gid >> 5;
    int lane = gid & 31;
    if (e >= EE) return;
    int r = ei[e], c = ei[EE + e];
    float norm = dis[r] * dis[c];
    float4 v = ((const float4*)(xw + (size_t)r * GH))[lane];
    float* dst = agg + (size_t)c * GH + lane * 4;
    atomicAdd(dst + 0, v.x * norm);
    atomicAdd(dst + 1, v.y * norm);
    atomicAdd(dst + 2, v.z * norm);
    atomicAdd(dst + 3, v.w * norm);
}

// h = relu(agg + gcn_b); segment-sum into per-graph accumulators
__global__ void pool_kernel(const float* __restrict__ agg,
                            const float* __restrict__ gb,
                            const int* __restrict__ batch,
                            float* __restrict__ sums, float* __restrict__ cnt)
{
    int i = blockIdx.x * blockDim.x + threadIdx.x;  // N*GH threads
    int n = i >> 7, f = i & 127;
    float h = fmaxf(agg[i] + gb[f], 0.f);
    int g = batch[n];
    atomicAdd(&sums[(size_t)g * GH + f], h);
    if (f == 0) atomicAdd(&cnt[g], 1.0f);
}

// per-graph mean + MLP head (128 -> 64 relu -> 2)
__global__ __launch_bounds__(128) void head_kernel(
    const float* __restrict__ sums, const float* __restrict__ cnt,
    const float* __restrict__ w1, const float* __restrict__ b1,
    const float* __restrict__ w2, const float* __restrict__ b2,
    float* __restrict__ out)
{
    __shared__ float gr[GH];
    __shared__ float hid[GH / 2];
    int g = blockIdx.x, tid = threadIdx.x;
    float c = fmaxf(cnt[g], 1.0f);
    gr[tid] = sums[(size_t)g * GH + tid] / c;
    __syncthreads();
    if (tid < GH / 2) {
        float a = b1[tid];
        #pragma unroll 8
        for (int k = 0; k < GH; ++k) a += gr[k] * w1[k * (GH / 2) + tid];
        hid[tid] = fmaxf(a, 0.f);
    }
    __syncthreads();
    if (tid < 2) {
        float a = b2[tid];
        #pragma unroll 8
        for (int k = 0; k < GH / 2; ++k) a += hid[k] * w2[k * 2 + tid];
        out[(size_t)g * 2 + tid] = a;
    }
}

// ============================================================
extern "C" void kernel_launch(void* const* d_in, const int* in_sizes, int n_in,
                              void* d_out, int out_size, void* d_ws, size_t ws_size,
                              hipStream_t stream)
{
    const float* x      = (const float*)d_in[0];
    const int*   ei     = (const int*)  d_in[1];
    const int*   batch  = (const int*)  d_in[2];
    const float* conv_w = (const float*)d_in[3];
    const float* conv_b = (const float*)d_in[4];
    const float* lin1_w = (const float*)d_in[5];
    const float* lin1_b = (const float*)d_in[6];
    const float* gcn_w  = (const float*)d_in[7];
    const float* gcn_b  = (const float*)d_in[8];
    const float* mlp_w1 = (const float*)d_in[9];
    const float* mlp_b1 = (const float*)d_in[10];
    const float* mlp_w2 = (const float*)d_in[11];
    const float* mlp_b2 = (const float*)d_in[12];
    float* out = (float*)d_out;

    // workspace carve-up (256B aligned slabs)
    size_t off = 0;
    auto carve = [&](size_t bytes) {
        void* p = (char*)d_ws + off;
        off += (bytes + 255) & ~(size_t)255;
        return p;
    };
    unsigned short* pooled = (unsigned short*)carve((size_t)NN * FLAT * 2);
    unsigned short* feats  = (unsigned short*)carve((size_t)NN * FD * 2);
    float* xw   = (float*)carve((size_t)NN * GH * 4);
    float* agg  = (float*)carve((size_t)NN * GH * 4);
    float* deg  = (float*)carve((size_t)NN * 4);
    float* dis  = (float*)carve((size_t)NN * 4);
    float* sums = (float*)carve((size_t)GG * GH * 4);
    float* cnt  = (float*)carve((size_t)GG * 4);

    // 1) CNN: conv+relu+maxpool via implicit-GEMM WMMA
    conv_pool_kernel<<<NN, 512, 0, stream>>>(x, conv_w, conv_b, pooled);

    // 2) Linear1: [N,512]@[512,64] + relu  (WMMA)
    lin1_kernel<<<NN / 32, 256, 0, stream>>>(pooled, lin1_w, lin1_b, feats);

    // 3) GCN xw: [N,64]@[64,128]  (WMMA)
    gcnxw_kernel<<<NN / 32, 512, 0, stream>>>(feats, gcn_w, xw);

    // 4) degrees / normalization
    int initTot = GG * GH;  // 131072 >= NN, GG
    init_kernel<<<(initTot + 255) / 256, 256, 0, stream>>>(deg, sums, cnt);
    deg_edges_kernel<<<(EE + 255) / 256, 256, 0, stream>>>(ei, deg);
    dis_kernel<<<(NN + 255) / 256, 256, 0, stream>>>(deg, dis);

    // 5) message passing: self-loop init + edge scatter
    selfloop_kernel<<<(NN * GH) / 256, 256, 0, stream>>>(xw, dis, agg);
    scatter_kernel<<<(EE * 32) / 256, 256, 0, stream>>>(ei, xw, dis, agg);

    // 6) bias + relu + global mean pool (segment sums)
    pool_kernel<<<(NN * GH) / 256, 256, 0, stream>>>(agg, gcn_b, batch, sums, cnt);

    // 7) MLP head
    head_kernel<<<GG, 128, 0, stream>>>(sums, cnt, mlp_w1, mlp_b1,
                                        mlp_w2, mlp_b2, out);
}
